// LIG_84963043049645
// MI455X (gfx1250) — matrix operand order repaired and checked by reading, not compiled, not run on hardware
//
#include <hip/hip_runtime.h>

typedef __attribute__((ext_vector_type(16))) _Float16 v16h;
typedef __attribute__((ext_vector_type(8)))  float    v8f;

#define HH 256
#define WW 256
#define NG 4096
#define STAGE 1024
#define BLOCK 256
#define WAVES 8

// Output float offsets (tuple concatenated flat, all f32):
// render(3,H,W) | img(H,W,3) | wsum(H,W) | dx(H,W,3) | dy(H,W,3) | dxy(H,W,3)
#define OFF_RENDER 0
#define OFF_IMG    196608
#define OFF_WSUM   393216
#define OFF_DX     458752
#define OFF_DY     655360
#define OFF_DXY    851968

#define LOG2E 1.4426950408889634f
#define LN2   0.6931471805599453f

__device__ __forceinline__ float fexp2(float x) { return __builtin_amdgcn_exp2f(x); }
__device__ __forceinline__ float frcp(float x)  { return __builtin_amdgcn_rcpf(x); }
__device__ __forceinline__ float flog2(float x) { return __builtin_amdgcn_logf(x); }

__device__ __forceinline__ float softplus_f(float x) {
  // log(1+e^x) = ln2 * log2(1 + 2^(x*log2e))
  return LN2 * flog2(1.0f + fexp2(x * LOG2E));
}
__device__ __forceinline__ float sigmoid_f(float x) {
  return frcp(1.0f + fexp2(-x * LOG2E));
}

__device__ __forceinline__ v8f wmma_f32_16x16x32(v16h a, v16h b, v8f c) {
  return __builtin_amdgcn_wmma_f32_16x16x32_f16(
      /*neg_a=*/false, a, /*neg_b=*/false, b,
      /*c_mod=*/(short)0, c, /*reuse_a=*/false, /*reuse_b=*/false);
}

__global__ __launch_bounds__(BLOCK)
void LIG_splat_wmma(const float* __restrict__ means,
                    const float* __restrict__ chol,
                    const float* __restrict__ rgbl,
                    const float* __restrict__ opl,
                    float* __restrict__ out) {
  // Packed per-Gaussian params: one b128 + one b64 LDS load per Gaussian.
  __shared__ float4 sp0[STAGE];        // (mx, my, a, b)
  __shared__ float2 sp1[STAGE];        // (c, o)
  __shared__ float  scol[3][STAGE];    // sigmoid colors, SoA per channel

  const int tid  = threadIdx.x;
  const int lane = tid & 31;
  const int wave = tid >> 5;

  // Each wave owns one horizontal strip of 16 pixels.
  const int strip = blockIdx.x * WAVES + wave;      // 0..4095
  const int y     = strip >> 4;                     // 16 strips per row
  const int xbase = (strip & 15) << 4;

  const int mrow  = lane & 15;                      // A-matrix row (pixel)
  const int khalf = (lane >> 4) << 3;               // A K-half base: 0 or 8
  const int kb2   = (lane >> 4) << 4;               // unused: 0 or 16 (B's K base is khalf)
  const int ncol  = lane & 15;                      // B/C column

  const float px = (float)(xbase + mrow) + 0.5f;
  const float py = (float)y + 0.5f;

  // Branchless B-column fetch: v = bptr[k]*bmul + badd
  //   ncol 0..2 -> color channel; ncol==3 -> constant 1 (wsum); else 0.
  const float* bptr = &scol[(ncol < 3) ? ncol : 2][0];
  const float  bmul = (ncol < 3) ? 1.0f : 0.0f;
  const float  badd = (ncol == 3) ? 1.0f : 0.0f;

  v8f accW = {}, accX = {}, accY = {}, accXY = {};

  for (int s = 0; s < NG / STAGE; ++s) {
    const int gbase = s * STAGE;
    __syncthreads();   // protect LDS from previous stage's readers

    // Cooperative stage: load raw params, transform once, store packed SoA.
    #pragma unroll
    for (int j = 0; j < STAGE / BLOCK; ++j) {
      const int kl = j * BLOCK + tid;
      const int kg = gbase + kl;
      const float mx = means[2 * kg + 0];
      const float my = means[2 * kg + 1];
      const float c0 = chol[3 * kg + 0];
      const float c1 = chol[3 * kg + 1];
      const float c2 = chol[3 * kg + 2];
      const float L11 = softplus_f(c0);
      const float L21 = c1;
      const float L22 = softplus_f(c2);
      const float s11 = L11 * L11;
      const float s12 = L11 * L21;
      const float s22 = L21 * L21 + L22 * L22;
      const float invdet = frcp(s11 * s22 - s12 * s12);
      sp0[kl] = make_float4(mx, my, s22 * invdet, -s12 * invdet);
      sp1[kl] = make_float2(s11 * invdet, sigmoid_f(opl[kg]));
      scol[0][kl] = sigmoid_f(rgbl[3 * kg + 0]);
      scol[1][kl] = sigmoid_f(rgbl[3 * kg + 1]);
      scol[2][kl] = sigmoid_f(rgbl[3 * kg + 2]);
    }
    __syncthreads();

    // Hint next stage into cache while we crunch this one.
    if (s + 1 < NG / STAGE) {
      const int kn = (s + 1) * STAGE + (tid << 2);
      __builtin_prefetch(&means[2 * kn], 0, 1);
      __builtin_prefetch(&chol[3 * kn], 0, 1);
      __builtin_prefetch(&rgbl[3 * kn], 0, 1);
    }

    for (int k0 = 0; k0 < STAGE; k0 += 32) {
      // B: 32x16 f16, lane = column N, element e -> K = khalf + e + (e>=8 ? 8 : 0). Branchless.
      v16h Bm;
      #pragma unroll
      for (int e = 0; e < 16; ++e) {
        const int k = k0 + khalf + e + ((e >> 3) << 3);
        Bm[e] = (_Float16)fmaf(bptr[k], bmul, badd);
      }

      // A: per lane 16 Gaussians for pixel mrow; K = khalf + e + (e>=8 ? 8 : 0).
      v16h Aw, Ax, Ay, Axy;
      #pragma unroll
      for (int e = 0; e < 16; ++e) {
        const int k  = k0 + khalf + e + ((e >> 3) << 3);
        const float4 p0 = sp0[k];                 // mx my a b  (ds_load_b128)
        const float2 p1 = sp1[k];                 // c o        (ds_load_b64)
        const float dx = px - p0.x;
        const float dy = py - p0.y;
        const float gx = fmaf(p0.z, dx, p0.w * dy);
        const float gy = fmaf(p0.w, dx, p1.x * dy);
        const float q  = fmaf(dx, gx, dy * gy);
        const float w  = p1.y * fexp2(-0.5f * LOG2E * q);
        Aw[e]  = (_Float16)w;
        Ax[e]  = (_Float16)(-w * gx);
        Ay[e]  = (_Float16)(-w * gy);
        Axy[e] = (_Float16)(w * fmaf(gx, gy, -p0.w));
      }

      // Four 16x16x32 matrix MACs: k-reduction on the matrix pipe.
      accW  = wmma_f32_16x16x32(Aw,  Bm, accW);
      accX  = wmma_f32_16x16x32(Ax,  Bm, accX);
      accY  = wmma_f32_16x16x32(Ay,  Bm, accY);
      accXY = wmma_f32_16x16x32(Axy, Bm, accXY);
    }
  }

  // Writeback: C layout — lane L, VGPR r -> pixel (r + (L>=16?8:0)), column L&15.
  const int mbase = (lane >> 4) << 3;
  if (ncol < 4) {
    #pragma unroll
    for (int r = 0; r < 8; ++r) {
      const int m   = r + mbase;
      const int pix = y * WW + (xbase + m);
      if (ncol < 3) {
        const float vw = accW[r];
        out[OFF_RENDER + ncol * (HH * WW) + pix] = vw;     // render (C,H,W)
        out[OFF_IMG + pix * 3 + ncol]            = vw;     // img (H,W,3)
        out[OFF_DX  + pix * 3 + ncol]            = accX[r];
        out[OFF_DY  + pix * 3 + ncol]            = accY[r];
        out[OFF_DXY + pix * 3 + ncol]            = accXY[r];
      } else {
        out[OFF_WSUM + pix] = accW[r];                     // wsum (col of ones)
      }
    }
  }
}

extern "C" void kernel_launch(void* const* d_in, const int* in_sizes, int n_in,
                              void* d_out, int out_size, void* d_ws, size_t ws_size,
                              hipStream_t stream) {
  (void)in_sizes; (void)n_in; (void)out_size; (void)d_ws; (void)ws_size;
  const float* means = (const float*)d_in[0];
  const float* chol  = (const float*)d_in[1];
  const float* rgbl  = (const float*)d_in[2];
  const float* opl   = (const float*)d_in[3];
  float* out = (float*)d_out;

  const int nstrips = (HH * WW) / 16;               // 4096
  const int nblocks = nstrips / WAVES;              // 512
  LIG_splat_wmma<<<nblocks, BLOCK, 0, stream>>>(means, chol, rgbl, opl, out);
}
